// LayerFused_83837761618032
// MI455X (gfx1250) — matrix operand order repaired and checked by reading, not compiled, not run on hardware
//
#include <hip/hip_runtime.h>
#include <hip/hip_bf16.h>

typedef __attribute__((ext_vector_type(16))) _Float16 v16h;
typedef __attribute__((ext_vector_type(8)))  float    v8f;

#define N_ 128
#define I_ 256
#define O_ 256
#define P_ 16
#define NSPLIT 4
#define ICHUNK (I_ / NSPLIT)

#define SQRT2PI     2.5066282746310002f
#define INV_SQRT2PI 0.3989422804014327f
#define GJITTER     0.1f
#define HALF_LOG2E  0.7213475204444817f   // 0.5 * log2(e)

// ---------------------------------------------------------------------------
// Kernel 0: zero the output (harness poisons d_out; we accumulate atomically)
// ---------------------------------------------------------------------------
__global__ void zero_kernel(float* __restrict__ p, int n) {
    int idx = blockIdx.x * blockDim.x + threadIdx.x;
    if (idx < n) p[idx] = 0.0f;
}

// ---------------------------------------------------------------------------
// Kernel 1: per-(i,o) precompute.
// 16 lanes cooperate on one 16x16 matrix (16 matrices / 256-thread block).
// Produces: z (f32), Qinv (f16, B-fragment layout), w = Qinv*h (f32), scalars.
// ---------------------------------------------------------------------------
__global__ __launch_bounds__(256) void precompute_kernel(
    const float* __restrict__ z_param, const float* __restrict__ h,
    const float* __restrict__ l_param, const float* __restrict__ s_param,
    const float* __restrict__ jitter_param,
    float* __restrict__ ws_z, _Float16* __restrict__ ws_qinv,
    float* __restrict__ ws_w, float* __restrict__ ws_scal)
{
    __shared__ float Abuf[16][256];   // 16 matrices, 16x16 row-major (r*16+c)
    __shared__ float Zs[16][16];

    const int tid = threadIdx.x;
    const int m   = tid >> 4;         // matrix slot within block
    const int p   = tid & 15;         // this lane's column
    const int io  = blockIdx.x * 16 + m;
    float* M = &Abuf[m][0];

    const float zp  = tanhf(z_param[io * P_ + p]);
    const float l   = __expf(l_param[io]) + 0.2f;   // > 0, so |l| = l
    const float s   = __expf(s_param[io]) + 0.1f;
    const float jit = __expf(jitter_param[io]) + 0.01f;
    const float l2  = l * l;
    const float inv2l2 = 0.5f / l2;
    const float norm   = 1.0f / (l * SQRT2PI);
    const float noise  = jit * jit / (s * s * l * SQRT2PI);

    Zs[m][p] = zp;
    __syncthreads();

    // Build column p of Q_hh + noise*I
    #pragma unroll
    for (int r = 0; r < 16; ++r) {
        float dz = Zs[m][r] - zp;
        float q  = __expf(-dz * dz * inv2l2) * norm;
        if (r == p) q += noise;
        M[r * 16 + p] = q;
    }
    __syncthreads();

    // In-place Cholesky (lower triangle), column-per-lane right-looking
    for (int k = 0; k < 16; ++k) {
        float lkk = sqrtf(M[k * 16 + k]);
        float inv = 1.0f / lkk;
        float colv = 0.0f;
        if (p >= k) colv = M[p * 16 + k] * inv;   // L[p][k] (p==k -> lkk)
        __syncthreads();
        if (p >= k) M[p * 16 + k] = colv;
        __syncthreads();
        if (p > k) {
            const float ljk = colv;               // L[p][k]
            #pragma unroll
            for (int r = 0; r < 16; ++r)
                if (r >= p) M[r * 16 + p] -= M[r * 16 + k] * ljk;
        }
        __syncthreads();
    }

    // Forward substitution: column p of L^-1  (solve L x = e_p)
    float x[16];
    #pragma unroll
    for (int r = 0; r < 16; ++r) {
        float acc = (r == p) ? 1.0f : 0.0f;
        #pragma unroll
        for (int k = 0; k < r; ++k) acc -= M[r * 16 + k] * x[k];
        x[r] = acc / M[r * 16 + r];
    }
    __syncthreads();
    #pragma unroll
    for (int k = 0; k < 16; ++k) M[k * 16 + p] = x[k];   // M := Linv
    __syncthreads();

    // Qinv column p: qv[r] = sum_k Linv[k][p] * Linv[k][r]
    float qv[16];
    #pragma unroll
    for (int r = 0; r < 16; ++r) {
        float acc = 0.0f;
        #pragma unroll
        for (int k = 0; k < 16; ++k) acc += x[k] * M[k * 16 + r];
        qv[r] = acc;
    }

    // w[p] = sum_r Qinv[r][p] * h[r]   (Qinv symmetric)
    float wp = 0.0f;
    #pragma unroll
    for (int r = 0; r < 16; ++r) wp += qv[r] * h[io * P_ + r];

    // Stores. Qinv layout: ws_qinv[io*256 + c*16 + k] = Qinv[k][c]
    // => B fragment for lane N=c, half h: 8 contiguous f16 (16B) at k=8h..8h+7.
    ws_z[io * P_ + p] = zp;
    ws_w[io * P_ + p] = wp;
    v16h qh;
    #pragma unroll
    for (int r = 0; r < 16; ++r) qh[r] = (_Float16)qv[r];
    *(v16h*)(ws_qinv + (size_t)io * 256 + p * 16) = qh;

    if (p == 0) {
        ws_scal[io * 4 + 0] = l2;
        ws_scal[io * 4 + 1] = s * s * l;            // c3: t3 = c3 * rsqrt(l2+2xv)
        ws_scal[io * 4 + 2] = SQRT2PI * s * s * l;  // t4
        ws_scal[io * 4 + 3] = 0.0f;
    }
}

// ---------------------------------------------------------------------------
// Kernel 2: main fused kernel.
// Block = (o, i-split); 8 waves cover all N=128 rows (16 rows per wave).
// Per i: build q_xh tile, one v_wmma_f32_16x16x32_f16 (K padded to 32),
// accumulate mean (q.w), t5 (q.(Qinv q)) scaled by t4, and t3 in registers.
// ---------------------------------------------------------------------------
__global__ __launch_bounds__(256) void main_kernel(
    const float* __restrict__ x_mean, const float* __restrict__ x_var,
    const float* __restrict__ ws_z, const _Float16* __restrict__ ws_qinv,
    const float* __restrict__ ws_w, const float* __restrict__ ws_scal,
    float* __restrict__ out_mean, float* __restrict__ out_var)
{
    const int o     = blockIdx.x;          // 0..255
    const int i0    = blockIdx.y * ICHUNK; // i-split start
    const int tid   = threadIdx.x;
    const int wave  = tid >> 5;            // 0..7
    const int lane  = tid & 31;
    const int row   = lane & 15;           // M (A) / N (B,D) index
    const int half  = lane >> 4;           // 0: p=0..7 | 1: p=8..15
    const int n     = wave * 16 + row;     // global sample index

    __shared__ float qtile[8][16 * 17];    // per-wave q tile, pitch 17 (bank-safe)
    __shared__ float red[8][16];

    float acc_m = 0.0f, acc_t3 = 0.0f;
    float acc_d[8];
    #pragma unroll
    for (int r = 0; r < 8; ++r) acc_d[r] = 0.0f;

    #pragma unroll 2
    for (int ii = 0; ii < ICHUNK; ++ii) {
        const int i  = i0 + ii;
        const size_t io = (size_t)i * O_ + o;

        const float4 sc = *(const float4*)(ws_scal + io * 4);
        const float l2 = sc.x, c3 = sc.y, t4 = sc.z;

        const float xm = x_mean[n * I_ + i];
        const float xv = x_var [n * I_ + i];
        const float v  = xv + l2;
        // v >= l2 >= 0.04: raw v_rcp/v_rsq are safe (no denorm-range fixup needed)
        const float rv = __builtin_amdgcn_rcpf(v);
        const float rs = __builtin_amdgcn_rsqf(v) * INV_SQRT2PI;
        const float ke = rv * HALF_LOG2E;      // exp(-d^2/(2v)) = exp2(-d^2*ke)

        acc_t3 += c3 * __builtin_amdgcn_rsqf(l2 + 2.0f * xv); // same in both halves

        // 8 inducing points for this lane: p = half*8 + j
        const float4 z0 = *(const float4*)(ws_z + io * 16 + half * 8);
        const float4 z1 = *(const float4*)(ws_z + io * 16 + half * 8 + 4);
        const float zarr[8] = {z0.x, z0.y, z0.z, z0.w, z1.x, z1.y, z1.z, z1.w};

        float q[8];
        v16h af;
        #pragma unroll
        for (int j = 0; j < 8; ++j) {
            float d  = xm - zarr[j];
            q[j] = __builtin_amdgcn_exp2f(-d * d * ke) * rs;
            af[j]     = (_Float16)q[j];       // K = half*8+j  (A layout)
            af[8 + j] = (_Float16)0.0f;       // K = 16..31 zero padding
        }

        // stage tile for D-layout transpose read
        #pragma unroll
        for (int j = 0; j < 8; ++j)
            qtile[wave][row * 17 + half * 8 + j] = q[j];

        // B fragment: element j<8 holds K=half*8+j of column N=row; K>=16 zero
        v16h bf = {};
        #pragma unroll
        for (int j = 0; j < 8; ++j)
            bf[j] = ws_qinv[io * 256 + row * 16 + half * 8 + j];

        v8f c = {};
        v8f d = __builtin_amdgcn_wmma_f32_16x16x32_f16(
            false, af, false, bf, (short)0, c, false, false);

        __builtin_amdgcn_wave_barrier();

        // t5 accumulation: D[r] = Y[M=r+8*half][N=row]; need q[M][row]
        #pragma unroll
        for (int r = 0; r < 8; ++r) {
            float qd = qtile[wave][(r + 8 * half) * 17 + row];
            acc_d[r] += t4 * qd * d[r];
        }

        // mean: q . w over this lane's 8 p's
        const float4 w0 = *(const float4*)(ws_w + io * 16 + half * 8);
        const float4 w1 = *(const float4*)(ws_w + io * 16 + half * 8 + 4);
        const float warr[8] = {w0.x, w0.y, w0.z, w0.w, w1.x, w1.y, w1.z, w1.w};
        #pragma unroll
        for (int j = 0; j < 8; ++j) acc_m += q[j] * warr[j];

        __builtin_amdgcn_wave_barrier();   // before next iter overwrites qtile
    }

    // combine the two half-lane partial mean sums (p 0..7 + p 8..15)
    const float m_tot = acc_m + __shfl_xor(acc_m, 16, 32);

    // reduce t5 over the 16 N-lanes within each half
    #pragma unroll
    for (int r = 0; r < 8; ++r) {
        acc_d[r] += __shfl_xor(acc_d[r], 1, 32);
        acc_d[r] += __shfl_xor(acc_d[r], 2, 32);
        acc_d[r] += __shfl_xor(acc_d[r], 4, 32);
        acc_d[r] += __shfl_xor(acc_d[r], 8, 32);
    }
    if (lane == 0) {
        #pragma unroll
        for (int r = 0; r < 8; ++r) red[wave][r] = acc_d[r];        // M=0..7
    }
    if (lane == 16) {
        #pragma unroll
        for (int r = 0; r < 8; ++r) red[wave][8 + r] = acc_d[r];    // M=8..15
    }
    __builtin_amdgcn_wave_barrier();

    if (half == 0) {
        const float t5sum = red[wave][row];
        const float var_p = acc_t3 - t5sum + (float)ICHUNK * GJITTER;
        atomicAdd(&out_mean[n * O_ + o], m_tot);
        atomicAdd(&out_var [n * O_ + o], var_p);
    }
}

// ---------------------------------------------------------------------------
extern "C" void kernel_launch(void* const* d_in, const int* in_sizes, int n_in,
                              void* d_out, int out_size, void* d_ws, size_t ws_size,
                              hipStream_t stream) {
    const float* x_mean  = (const float*)d_in[0];
    const float* x_var   = (const float*)d_in[1];
    const float* z_param = (const float*)d_in[2];
    const float* h       = (const float*)d_in[3];
    const float* l_param = (const float*)d_in[4];
    const float* s_param = (const float*)d_in[5];
    const float* jitter  = (const float*)d_in[6];
    float* out = (float*)d_out;

    const size_t IO = (size_t)I_ * O_;
    // Workspace carve-up (~41 MB total):
    _Float16* ws_qinv = (_Float16*)d_ws;                              // IO*256 f16 = 32 MB
    float*    ws_z    = (float*)((char*)d_ws + IO * 256 * sizeof(_Float16)); // 4 MB
    float*    ws_w    = ws_z + IO * 16;                               // 4 MB
    float*    ws_scal = ws_w + IO * 16;                               // 1 MB

    zero_kernel<<<(2 * N_ * O_ + 255) / 256, 256, 0, stream>>>(out, 2 * N_ * O_);
    precompute_kernel<<<IO / 16, 256, 0, stream>>>(
        z_param, h, l_param, s_param, jitter, ws_z, ws_qinv, ws_w, ws_scal);
    main_kernel<<<dim3(O_, NSPLIT), 256, 0, stream>>>(
        x_mean, x_var, ws_z, ws_qinv, ws_w, ws_scal, out, out + N_ * O_);
}
